// HyperNetLinearLayer_6399501271812
// MI455X (gfx1250) — matrix operand order repaired and checked by reading, not compiled, not run on hardware
//
#include <hip/hip_runtime.h>
#include <hip/hip_bf16.h>

// HyperNetLinearLayer fused as ONE GEMM:
//   out[b,o] = sum_k A[b,k] * Wbt[o,k] + base_bias[o] + b2[65536+o]
// where k < 32768:  A = x[b,i]*h[b,hh], Wbt = w2[hh, i*256+o]   (k = i*128+hh)
//       k in [32768,33024): A = x[b,k-32768], Wbt = base_weight + b2-head
//       k in [33024,33152): A = h[b,k-33024], Wbt = w2[.., 65536+o]
//       k in [33152,33792): zero padding
// A is built on the fly in LDS (f16); Wbt is repacked once into d_ws (f16).
// f32 accumulation via v_wmma_f32_16x16x32_f16.
// Workspace: 256*33792*2 + 512*128*4 bytes = ~17.6 MB.

typedef __attribute__((ext_vector_type(16))) _Float16 v16h;
typedef __attribute__((ext_vector_type(8)))  _Float16 v8h;
typedef __attribute__((ext_vector_type(8)))  float    v8f;

#define KP        33792      // padded K' (33 chunks of 1024)
#define CHUNK     1024
#define A_STRIDE  1032       // 1024 + 8 halves padding (bank-conflict avoidance)
#define W2_ROW    65792      // INPUT_DIM*OUTPUT_DIM/.. = 256*256 + 256

// ---------------------------------------------------------------- h = relu(cond@w1+b1)
__global__ __launch_bounds__(128)
void h_kernel(const float* __restrict__ cond, const float* __restrict__ w1,
              const float* __restrict__ b1, float* __restrict__ h)
{
    const int b = blockIdx.x;          // 512
    const int j = threadIdx.x;         // 128
    float acc = b1[j];
    for (int k = 0; k < 128; ++k)
        acc += cond[b * 128 + k] * w1[k * 128 + j];   // w1 coalesced across threads
    h[b * 128 + j] = fmaxf(acc, 0.0f);
}

// ---------------------------------------------------------------- Wbt[o][i*128+hh] = f16(w2[hh][i*256+o])
__global__ __launch_bounds__(256)
void pack_kernel(const float* __restrict__ w2, _Float16* __restrict__ Wbt)
{
    __shared__ float tile[16][17];
    const int tx = threadIdx.x, ty = threadIdx.y;    // 16x16
    const int o0  = blockIdx.x * 16;                 // 16 tiles
    const int hh0 = blockIdx.y * 16;                 // 8 tiles
    const int i   = blockIdx.z;                      // 256
    tile[ty][tx] = w2[(size_t)(hh0 + ty) * W2_ROW + i * 256 + o0 + tx];  // coalesced read
    __syncthreads();
    Wbt[(size_t)(o0 + ty) * KP + i * 128 + hh0 + tx] = (_Float16)tile[tx][ty]; // coalesced write
}

// ---------------------------------------------------------------- tail columns + zero pad
__global__ __launch_bounds__(256)
void pack_tail_kernel(const float* __restrict__ base_weight, const float* __restrict__ b2,
                      const float* __restrict__ w2, _Float16* __restrict__ Wbt)
{
    const int o = blockIdx.x;        // 256
    const int t = threadIdx.x;       // 256
    _Float16* row = Wbt + (size_t)o * KP;
    // x-tail: base_weight + head of b2 (covers x@base_weight and x@b2-as-matrix)
    row[32768 + t] = (_Float16)(base_weight[t * 256 + o] + b2[t * 256 + o]);
    // h-tail: last 256 columns of w2 (covers bias_delta = h @ w2[:, 65536:])
    if (t < 128)
        row[33024 + t] = (_Float16)w2[(size_t)t * W2_ROW + 65536 + o];
    // zero padding up to KP
    for (int k = 33152 + t; k < KP; k += 256)
        row[k] = (_Float16)0.0f;
}

// ---------------------------------------------------------------- fused big GEMM
__global__ __launch_bounds__(128)
void gemm_kernel(const float* __restrict__ x, const float* __restrict__ h,
                 const _Float16* __restrict__ Wbt, const float* __restrict__ base_bias,
                 const float* __restrict__ b2, float* __restrict__ out)
{
    __shared__ _Float16 A_lds[16 * A_STRIDE];   // 33 KB: 16 b-rows x 1024 k (f16, padded)
    __shared__ float    xs[16 * 256];           // 16 KB
    __shared__ float    hs[16 * 128];           //  8 KB

    const int t     = threadIdx.x;          // 128 threads = 4 waves (wave32)
    const int lane  = t & 31;
    const int wave  = t >> 5;
    const int b0    = blockIdx.x * 16;              // 32 m-tiles
    const int n0    = blockIdx.y * 64 + wave * 16;  // 4 n-groups x 4 waves -> o in [0,256)
    const int hiSel = (lane >> 4) & 1;              // upper half-wave selector
    const int m     = lane & 15;

    // Stage this block's x and h rows (f32) into LDS once.
    for (int idx = t; idx < 16 * 256; idx += 128) {
        int bl = idx >> 8, c = idx & 255;
        xs[idx] = x[(b0 + bl) * 256 + c];
    }
    for (int idx = t; idx < 16 * 128; idx += 128) {
        int bl = idx >> 7, c = idx & 127;
        hs[idx] = h[(b0 + bl) * 128 + c];
    }
    __syncthreads();

    v8f acc = {};   // 16x16 f32 accumulator (8 VGPRs)

    for (int c = 0; c < 33; ++c) {
        // ---- build A chunk (16 x 1024 f16) in LDS: packed pairs (one u32 store) ----
        if (c < 32) {
            const int i0 = c * 8;   // 8 i-values per chunk (8*128 = 1024 k)
            for (int idx = t; idx < 16 * 512; idx += 128) {
                int bl = idx >> 9;
                int kk = (idx & 511) << 1;
                int ii = i0 + (kk >> 7);
                int hh = kk & 127;
                float xv = xs[bl * 256 + ii];
                union { _Float16 f[2]; unsigned u; } pk;
                pk.f[0] = (_Float16)(xv * hs[bl * 128 + hh]);
                pk.f[1] = (_Float16)(xv * hs[bl * 128 + hh + 1]);
                *(unsigned*)&A_lds[bl * A_STRIDE + kk] = pk.u;
            }
        } else {    // tail chunk: k' in [0,256)->x, [256,384)->h, else 0
            for (int idx = t; idx < 16 * 512; idx += 128) {
                int bl = idx >> 9;
                int kk = (idx & 511) << 1;
                union { _Float16 f[2]; unsigned u; } pk;
                #pragma unroll
                for (int p = 0; p < 2; ++p) {
                    int k1 = kk + p;
                    float v = (k1 < 256) ? xs[bl * 256 + k1]
                            : (k1 < 384) ? hs[bl * 128 + (k1 - 256)]
                            : 0.0f;
                    pk.f[p] = (_Float16)v;
                }
                *(unsigned*)&A_lds[bl * A_STRIDE + kk] = pk.u;
            }
        }
        __syncthreads();

        // ---- WMMA over this chunk: 32 steps of K=32 ----
        const _Float16* brow = Wbt + (size_t)(n0 + m) * KP + (size_t)c * CHUNK + hiSel * 8;
        for (int kb = 0; kb < CHUNK; kb += 32) {
            union { v16h v; v8h half8[2]; } af, bf;
            // A frag (16x32 f16): lanes 0-15 hold K {kb+0..7, kb+16..23}, lanes 16-31 {+8..15, +24..31}
            const _Float16* ap = &A_lds[m * A_STRIDE + kb + hiSel * 8];
            af.half8[0] = *(const v8h*)(ap);         // ds_load_b128
            af.half8[1] = *(const v8h*)(ap + 16);    // ds_load_b128
            // B frag (32x16 f16): lane n=lane&15; lanes 0-15 K {kb+0..7, kb+16..23}, lanes 16-31 {+8..15, +24..31}
            bf.half8[0] = *(const v8h*)(brow + kb);      // global_load_b128
            bf.half8[1] = *(const v8h*)(brow + kb + 16); // global_load_b128
            acc = __builtin_amdgcn_wmma_f32_16x16x32_f16(
                false, af.v, false, bf.v, (short)0, acc, false, false);
        }
        __syncthreads();
    }

    // ---- epilogue: biases + store (C layout: VGPR r -> M = r + 8*hiSel, N = lane&15) ----
    const int nn = n0 + m;
    const float bias = base_bias[nn] + b2[65536 + nn];
    #pragma unroll
    for (int r = 0; r < 8; ++r) {
        int mm = r + hiSel * 8;
        out[(size_t)(b0 + mm) * 256 + nn] = acc[r] + bias;
    }
}

// ---------------------------------------------------------------- launch
extern "C" void kernel_launch(void* const* d_in, const int* in_sizes, int n_in,
                              void* d_out, int out_size, void* d_ws, size_t ws_size,
                              hipStream_t stream)
{
    const float* x           = (const float*)d_in[0];   // (512,256)
    const float* cond        = (const float*)d_in[1];   // (512,128)
    const float* base_weight = (const float*)d_in[2];   // (256,256)
    const float* base_bias   = (const float*)d_in[3];   // (256,)
    const float* w1          = (const float*)d_in[4];   // (128,128)
    const float* b1          = (const float*)d_in[5];   // (128,)
    const float* w2          = (const float*)d_in[6];   // (128,65792)
    const float* b2          = (const float*)d_in[7];   // (65792,)
    float*       out         = (float*)d_out;           // (512,256)

    _Float16* Wbt  = (_Float16*)d_ws;                                   // 256*KP halves
    float*    h_ws = (float*)((char*)d_ws + (size_t)256 * KP * 2);      // 512*128 f32

    h_kernel<<<512, 128, 0, stream>>>(cond, w1, b1, h_ws);
    pack_kernel<<<dim3(16, 8, 256), dim3(16, 16), 0, stream>>>(w2, Wbt);
    pack_tail_kernel<<<256, 256, 0, stream>>>(base_weight, b2, w2, Wbt);
    gemm_kernel<<<dim3(32, 4), 128, 0, stream>>>(x, h_ws, Wbt, base_bias, b2, out);
}
